// EncoderSTB_41764261987147
// MI455X (gfx1250) — compile-verified, matrix operand order checked by reading, not run on hardware
//
#include <hip/hip_runtime.h>

#define DIMC 256
#define HEADS 8
#define HDIM 32
#define NTOK 4096
#define TOPK 128
#define HID 1024
#define SCALE 0.17677669529663687f /* 32^-0.5 */
#define EPS 1e-5f

typedef __bf16 bf16;
typedef __attribute__((ext_vector_type(16))) __bf16 v16bf;
typedef __attribute__((ext_vector_type(8))) float v8f;
typedef __attribute__((ext_vector_type(4))) unsigned v4u;
typedef __attribute__((ext_vector_type(8))) unsigned v8u;

static __device__ inline v8f zero8() {
  v8f z;
#pragma unroll
  for (int i = 0; i < 8; ++i) z[i] = 0.f;
  return z;
}

static __device__ inline v8f wmma_bf16(v16bf a, v16bf b, v8f c) {
  return __builtin_amdgcn_wmma_f32_16x16x32_bf16(false, a, false, b, (short)0, c,
                                                 false, false);
}

// A-style fragment: element(m,k) = base[m*ld + k]; k-pairs contiguous -> b32 loads
// with constant offsets. (Also B stored column-major: element(k,n)=base[n*ld+k].)
static __device__ inline v16bf load_frag_contig(const bf16* base, int ld) {
  const int lane = threadIdx.x & 31;
  const int m = lane & 15;
  const int kh = (lane >> 4) << 3;
  const bf16* p = base + (size_t)m * ld + kh;
  v16bf f;
#pragma unroll
  for (int r = 0; r < 8; ++r) {
    int k = ((r & 4) << 2) + ((r & 3) << 1);
    union { unsigned u; bf16 h[2]; } t;
    t.u = *(const unsigned*)(p + k);
    f[2 * r] = t.h[0];
    f[2 * r + 1] = t.h[1];
  }
  return f;
}

// Pair-interleaved packed operand: pair (k,k+1) lives at base + k*KS + 2n,
// so each lane issues 8 aligned b32 loads at compile-time-constant offsets.
template <int KS>
static __device__ inline v16bf load_frag_packed(const bf16* base) {
  const int lane = threadIdx.x & 31;
  const int n = lane & 15;
  const int kh = (lane >> 4) << 3;
  const bf16* p = base + (size_t)kh * KS + 2 * n;
  v16bf f;
#pragma unroll
  for (int r = 0; r < 8; ++r) {
    int k = ((r & 4) << 2) + ((r & 3) << 1);
    union { unsigned u; bf16 h[2]; } t;
    t.u = *(const unsigned*)(p + (size_t)k * KS);
    f[2 * r] = t.h[0];
    f[2 * r + 1] = t.h[1];
  }
  return f;
}

// ---------------- weight conversion + pair-interleave packing ----------------
// Bp[(k & ~1)*256 + 2n + (k & 1)] = B[k][n]   (N fixed at 256)
__global__ void packw_kernel(const float* __restrict__ in, bf16* __restrict__ out,
                             int total) {
  int i = blockIdx.x * 256 + threadIdx.x;
  if (i < total) {
    int k = i >> 8, n = i & 255;
    out[((k & ~1) << 8) + (n << 1) + (k & 1)] = (bf16)in[i];
  }
}

__global__ void packkv_kernel(const float* __restrict__ kv, bf16* __restrict__ wk,
                              bf16* __restrict__ wv) {
  int i = blockIdx.x * 256 + threadIdx.x; // 256*512
  int k = i >> 9, c = i & 511;
  int n = c & 255;
  int o = ((k & ~1) << 8) + (n << 1) + (k & 1);
  bf16 v = (bf16)kv[i];
  if (c < 256) wk[o] = v;
  else         wv[o] = v;
}

// c1_w is (O=256, I=2048); GEMM B matrix is (k=I, n=O)
__global__ void packc1_kernel(const float* __restrict__ c1w, bf16* __restrict__ out) {
  int i = blockIdx.x * 256 + threadIdx.x; // 256*2048
  int o = i >> 11, ic = i & 2047;
  out[((ic & ~1) << 8) + (o << 1) + (ic & 1)] = (bf16)c1w[i];
}

// V repacked per head for the PV WMMA B operand: KS = 32
__global__ void packv_kernel(const bf16* __restrict__ vbf, bf16* __restrict__ vpk) {
  int i = blockIdx.x * 256 + threadIdx.x; // 4096*256
  int key = i >> 8, c = i & 255;
  int h = c >> 5, d = c & 31;
  vpk[(size_t)h * (NTOK * HDIM) + ((key & ~1) << 5) + (d << 1) + (key & 1)] = vbf[i];
}

// ---------------- layernorm ----------------
__global__ __launch_bounds__(256) void ln_kernel(const float* __restrict__ x,
                                                 const float* __restrict__ g,
                                                 const float* __restrict__ b,
                                                 bf16* __restrict__ out_bf,
                                                 float* __restrict__ out_tf) {
  __shared__ float red[256];
  const int n = blockIdx.x, c = threadIdx.x;
  float v = x[(size_t)n * DIMC + c];
  red[c] = v;
  __syncthreads();
#pragma unroll
  for (int s = 128; s > 0; s >>= 1) {
    if (c < s) red[c] += red[c + s];
    __syncthreads();
  }
  float mu = red[0] * (1.0f / DIMC);
  __syncthreads();
  float d = v - mu;
  red[c] = d * d;
  __syncthreads();
#pragma unroll
  for (int s = 128; s > 0; s >>= 1) {
    if (c < s) red[c] += red[c + s];
    __syncthreads();
  }
  float var = red[0] * (1.0f / DIMC);
  float y = d * rsqrtf(var + EPS) * g[c] + b[c];
  if (out_bf) out_bf[(size_t)n * DIMC + c] = (bf16)y;
  if (out_tf) out_tf[(size_t)c * NTOK + n] = y; // channel-major for conv
}

// ---------------- WMMA GEMM (Nc = 256 fixed), one wave per 16x16 tile ----------
// APACK=0: A row-major bf16, lda elements. APACK=1: A pair-interleaved KS=4096.
template <int APACK>
__global__ __launch_bounds__(128) void gemm16_kernel(
    const bf16* __restrict__ A, int lda, const bf16* __restrict__ Bp,
    const float* __restrict__ bias, const float* __restrict__ resid,
    float* __restrict__ outf, bf16* __restrict__ outb, int nTiles, int Ksteps) {
  const int wave = blockIdx.x * (blockDim.x >> 5) + (threadIdx.x >> 5);
  if (wave >= nTiles) return;
  const int tm = wave >> 4, tn = wave & 15; // 16 tiles along N (256/16)
  const int lane = threadIdx.x & 31;
  v8f acc = zero8();
  for (int ks = 0; ks < Ksteps; ++ks) {
    if (ks + 1 < Ksteps) {
      __builtin_prefetch(Bp + (size_t)((ks + 1) * 32) * 256 + tn * 32, 0, 1);
      if (!APACK)
        __builtin_prefetch(A + (size_t)(tm * 16) * lda + (ks + 1) * 32, 0, 1);
    }
    v16bf a = APACK ? load_frag_packed<4096>(A + (size_t)(ks * 32) * 4096 + tm * 32)
                    : load_frag_contig(A + (size_t)(tm * 16) * lda + ks * 32, lda);
    v16bf b = load_frag_packed<256>(Bp + (size_t)(ks * 32) * 256 + tn * 32);
    acc = wmma_bf16(a, b, acc);
  }
  const int half = lane >> 4;
  const int col = tn * 16 + (lane & 15);
  const float bv = bias ? bias[col] : 0.f;
#pragma unroll
  for (int r = 0; r < 8; ++r) {
    int row = tm * 16 + r + (half << 3);
    size_t o = (size_t)row * DIMC + col;
    float v = acc[r] + bv;
    if (resid) v += resid[o];
    if (outf) outf[o] = v;
    else      outb[o] = (bf16)v;
  }
}

// ---------------- per-head column sums of V ----------------
__global__ __launch_bounds__(256) void vsum_kernel(const bf16* __restrict__ vbf,
                                                   float* __restrict__ vs) {
  int c = threadIdx.x;
  float s = 0.f;
  for (int n = 0; n < NTOK; ++n) s += (float)vbf[(size_t)n * DIMC + c];
  vs[c] = s;
}

// ---------------- sparse attention ----------------
// One block per (16 queries, head). Exact top-128 via 4-pass radix select with
// WMMA-recomputed scores (compute is free vs 537MB of HBM score traffic).
// Masked softmax closed form:
//   out = [ sum_sel (e_i - e0) v_i + e0 * Vsum ] / (sum_sel e_i + (N-K) e0).
// Q tile is staged into LDS by the Tensor Data Mover (TDM).
__global__ __launch_bounds__(256) void attn_kernel(
    const bf16* __restrict__ qbf, const bf16* __restrict__ kbf,
    const bf16* __restrict__ vpk, const float* __restrict__ vs,
    bf16* __restrict__ attnout) {
  __shared__ unsigned hist[16][256];
  __shared__ unsigned rowmaxu[16];
  __shared__ unsigned prefix[16];
  __shared__ int remaining[16];
  __shared__ float Dsel[16];
  __shared__ float m_s[16];
  __shared__ float e0_s[16];
  __shared__ __attribute__((aligned(16))) float outacc[16][32];
  __shared__ __attribute__((aligned(16))) bf16 Ptile[8][16][32];
  __shared__ __attribute__((aligned(16))) bf16 Qs[16 * 32];

  const int h = blockIdx.y;
  const int qt = blockIdx.x;
  const int tid = threadIdx.x;
  const int w = tid >> 5;
  const int lane = tid & 31;
  const int half = lane >> 4;

  // ---- TDM: DMA the 16x32 bf16 Q tile (row stride 256) into LDS ----
  if (tid == 0) {
    unsigned long long ga =
        (unsigned long long)(size_t)(qbf + (size_t)(qt * 16) * DIMC + h * HDIM);
    unsigned ldsoff = (unsigned)(size_t)(&Qs[0]); // LDS aperture: offset in low 32b
    v4u g0;
    g0[0] = 1u;                                   // count=1, user descriptor
    g0[1] = ldsoff;                               // lds_addr
    g0[2] = (unsigned)ga;                         // global_addr[31:0]
    g0[3] = (unsigned)((ga >> 32) & 0x01FFFFFFu) | (2u << 30); // addr[56:32], type=2
    v8u g1;
    g1[0] = 1u << 16;     // workgroup_mask=0, data_size=1 (2 bytes)
    g1[1] = 32u << 16;    // tensor_dim0 = 32
    g1[2] = 16u << 16;    // tensor_dim0 hi=0, tensor_dim1 = 16
    g1[3] = 32u << 16;    // tensor_dim1 hi=0, tile_dim0 = 32
    g1[4] = 16u;          // tile_dim1 = 16, tile_dim2 = 0
    g1[5] = 256u;         // tensor_dim0_stride = 256 (elements)
    g1[6] = 0u;
    g1[7] = 0u;
    asm volatile("tensor_load_to_lds %0, %1" ::"s"(g0), "s"(g1) : "memory");
    __builtin_amdgcn_s_wait_tensorcnt(0);
  }

  unsigned* hp = &hist[0][0];
  for (int i = tid; i < 16 * 256; i += 256) hp[i] = 0u;
  if (tid < 16) {
    rowmaxu[tid] = 0u;
    prefix[tid] = 0u;
    remaining[tid] = TOPK;
    Dsel[tid] = 0.f;
  }
  for (int i = tid; i < 16 * 32; i += 256) (&outacc[0][0])[i] = 0.f;
  __syncthreads();

  const v16bf qfrag = load_frag_contig(&Qs[0], 32);
  const bf16* kbase = kbf + h * HDIM;
  const bf16* vbase = vpk + (size_t)h * (NTOK * HDIM);

  // ---- radix select, MSB byte first ----
  for (int byte = 3; byte >= 0; --byte) {
    const int shift = byte << 3;
    for (int kt = w; kt < NTOK / 16; kt += 8) {
      v16bf kf = load_frag_contig(kbase + (size_t)(kt * 16) * DIMC, DIMC);
      v8f s = wmma_bf16(qfrag, kf, zero8());
#pragma unroll
      for (int r = 0; r < 8; ++r) {
        float a = s[r] * SCALE;
        unsigned bits = __float_as_uint(a);
        unsigned u = (bits & 0x80000000u) ? ~bits : (bits | 0x80000000u);
        int row = r + (half << 3);
        if (byte == 3) {
          atomicMax(&rowmaxu[row], u);
          atomicAdd(&hist[row][u >> 24], 1u);
        } else if ((u >> (shift + 8)) == (prefix[row] >> (shift + 8))) {
          atomicAdd(&hist[row][(u >> shift) & 255u], 1u);
        }
      }
    }
    __syncthreads();
    if (tid < 16) {
      unsigned rem = (unsigned)remaining[tid];
      unsigned cum = 0;
      int bin = 0;
      for (int bb = 255; bb >= 0; --bb) {
        unsigned cnt = hist[tid][bb];
        if (cum + cnt >= rem) { bin = bb; break; }
        cum += cnt;
      }
      prefix[tid] |= ((unsigned)bin) << shift;
      remaining[tid] = (int)(rem - cum);
    }
    __syncthreads();
    for (int i = tid; i < 16 * 256; i += 256) hp[i] = 0u;
    __syncthreads();
  }

  if (tid < 16) {
    unsigned u = rowmaxu[tid];
    unsigned bits = (u & 0x80000000u) ? (u & 0x7fffffffu) : ~u;
    float mx = __uint_as_float(bits);
    float m = fmaxf(0.f, mx); // masked zeros participate in the softmax max
    m_s[tid] = m;
    e0_s[tid] = expf(-m);
  }
  __syncthreads();

  // ---- final pass: build sparse P tiles, dense WMMA P @ V ----
  v8f acc0 = zero8(), acc1 = zero8();
  for (int kb = w; kb < NTOK / 32; kb += 8) {
    const int key0 = kb << 5;
#pragma unroll
    for (int sub = 0; sub < 2; ++sub) {
      int kt = (kb << 1) + sub;
      v16bf kf = load_frag_contig(kbase + (size_t)(kt * 16) * DIMC, DIMC);
      v8f s = wmma_bf16(qfrag, kf, zero8());
#pragma unroll
      for (int r = 0; r < 8; ++r) {
        int row = r + (half << 3);
        float a = s[r] * SCALE;
        unsigned bits = __float_as_uint(a);
        unsigned u = (bits & 0x80000000u) ? ~bits : (bits | 0x80000000u);
        bool sel = (u >= prefix[row]);
        float e = sel ? expf(a - m_s[row]) : 0.f;
        float pv = sel ? (e - e0_s[row]) : 0.f;
        Ptile[w][row][(sub << 4) + (lane & 15)] = (bf16)pv;
        float es = e;
        es += __shfl_xor(es, 1, 16);
        es += __shfl_xor(es, 2, 16);
        es += __shfl_xor(es, 4, 16);
        es += __shfl_xor(es, 8, 16);
        if ((lane & 15) == 0) atomicAdd(&Dsel[row], es);
      }
    }
    v16bf pf = load_frag_contig(&Ptile[w][0][0], 32);
    v16bf v0 = load_frag_packed<32>(vbase + (size_t)key0 * HDIM);
    v16bf v1 = load_frag_packed<32>(vbase + (size_t)key0 * HDIM + 32);
    acc0 = wmma_bf16(pf, v0, acc0);
    acc1 = wmma_bf16(pf, v1, acc1);
  }
#pragma unroll
  for (int r = 0; r < 8; ++r) {
    int row = r + (half << 3);
    atomicAdd(&outacc[row][lane & 15], acc0[r]);
    atomicAdd(&outacc[row][16 + (lane & 15)], acc1[r]);
  }
  __syncthreads();

  for (int i = tid; i < 16 * 32; i += 256) {
    int row = i >> 5, d = i & 31;
    float e0 = e0_s[row];
    float D = Dsel[row] + (float)(NTOK - TOPK) * e0;
    float val = (outacc[row][d] + e0 * vs[h * HDIM + d]) / D;
    attnout[(size_t)(qt * 16 + row) * DIMC + h * HDIM + d] = (bf16)val;
  }
}

// ---------------- MSFN depthwise convs -> packed xc (KS=4096 A operand) -------
__global__ __launch_bounds__(256) void msfn_conv_kernel(
    const float* __restrict__ xt, const float* __restrict__ c3w,
    const float* __restrict__ c3b, const float* __restrict__ c5w,
    const float* __restrict__ c5b, bf16* __restrict__ xc) {
  int idx = blockIdx.x * 256 + threadIdx.x; // HID * NTOK threads
  int oc = idx >> 12;
  int pp = idx & 4095;
  int g = oc >> 2; // groups=256, 4 out-channels per input channel
  int yy = pp >> 6, xx = pp & 63;
  const float* src = xt + (size_t)g * NTOK;
  float a3 = c3b[oc], a5 = c5b[oc];
#pragma unroll
  for (int dy = -2; dy <= 2; ++dy) {
#pragma unroll
    for (int dx = -2; dx <= 2; ++dx) {
      int y2 = yy + dy, x2 = xx + dx;
      float v = (y2 >= 0 && y2 < 64 && x2 >= 0 && x2 < 64) ? src[y2 * 64 + x2] : 0.f;
      a5 += v * c5w[oc * 25 + (dy + 2) * 5 + (dx + 2)];
      if (dy >= -1 && dy <= 1 && dx >= -1 && dx <= 1)
        a3 += v * c3w[oc * 9 + (dy + 1) * 3 + (dx + 1)];
    }
  }
  int k3 = oc, k5 = HID + oc;
  xc[((size_t)(k3 & ~1) << 12) + (pp << 1) + (k3 & 1)] = (bf16)a3;
  xc[((size_t)(k5 & ~1) << 12) + (pp << 1) + (k5 & 1)] = (bf16)a5;
}

extern "C" void kernel_launch(void* const* d_in, const int* in_sizes, int n_in,
                              void* d_out, int out_size, void* d_ws, size_t ws_size,
                              hipStream_t stream) {
  const float* x      = (const float*)d_in[0];
  const float* ln1_g  = (const float*)d_in[1];
  const float* ln1_b  = (const float*)d_in[2];
  const float* q_w    = (const float*)d_in[3];
  const float* q_b    = (const float*)d_in[4];
  const float* kv_w   = (const float*)d_in[5];
  const float* kv_b   = (const float*)d_in[6];
  const float* proj_w = (const float*)d_in[7];
  const float* proj_b = (const float*)d_in[8];
  const float* ln2_g  = (const float*)d_in[9];
  const float* ln2_b  = (const float*)d_in[10];
  const float* c3_w   = (const float*)d_in[11];
  const float* c3_b   = (const float*)d_in[12];
  const float* c5_w   = (const float*)d_in[13];
  const float* c5_b   = (const float*)d_in[14];
  const float* c1_w   = (const float*)d_in[15];
  const float* c1_b   = (const float*)d_in[16];
  float* out = (float*)d_out;

  char* p = (char*)d_ws;
  auto take = [&](size_t bytes) {
    char* r = p;
    p += (bytes + 255) & ~(size_t)255;
    return r;
  };
  bf16* wq      = (bf16*)take((size_t)DIMC * DIMC * 2);
  bf16* wk      = (bf16*)take((size_t)DIMC * DIMC * 2);
  bf16* wv      = (bf16*)take((size_t)DIMC * DIMC * 2);
  bf16* wp      = (bf16*)take((size_t)DIMC * DIMC * 2);
  bf16* c1wT    = (bf16*)take((size_t)2 * HID * DIMC * 2);
  bf16* xn1     = (bf16*)take((size_t)NTOK * DIMC * 2);
  bf16* qbf     = (bf16*)take((size_t)NTOK * DIMC * 2);
  bf16* kbf     = (bf16*)take((size_t)NTOK * DIMC * 2);
  bf16* vbf     = (bf16*)take((size_t)NTOK * DIMC * 2);
  bf16* vpk     = (bf16*)take((size_t)NTOK * DIMC * 2);
  float* vs     = (float*)take((size_t)DIMC * 4);
  bf16* attnout = (bf16*)take((size_t)NTOK * DIMC * 2);
  float* x1     = (float*)take((size_t)NTOK * DIMC * 4);
  float* xn2t   = (float*)take((size_t)NTOK * DIMC * 4);
  bf16* xc      = (bf16*)take((size_t)2 * HID * NTOK * 2);

  // weight conversion into pair-interleaved layouts
  packw_kernel<<<256, 256, 0, stream>>>(q_w, wq, DIMC * DIMC);
  packkv_kernel<<<512, 256, 0, stream>>>(kv_w, wk, wv);
  packw_kernel<<<256, 256, 0, stream>>>(proj_w, wp, DIMC * DIMC);
  packc1_kernel<<<2048, 256, 0, stream>>>(c1_w, c1wT);

  // LN1 -> bf16
  ln_kernel<<<NTOK, 256, 0, stream>>>(x, ln1_g, ln1_b, xn1, nullptr);

  // q/k/v projections (4096x256x256 each), head-major columns
  const int tiles = (NTOK / 16) * (DIMC / 16); // 4096
  gemm16_kernel<0><<<tiles / 4, 128, 0, stream>>>(xn1, DIMC, wq, q_b, nullptr,
                                                  nullptr, qbf, tiles, DIMC / 32);
  gemm16_kernel<0><<<tiles / 4, 128, 0, stream>>>(xn1, DIMC, wk, kv_b, nullptr,
                                                  nullptr, kbf, tiles, DIMC / 32);
  gemm16_kernel<0><<<tiles / 4, 128, 0, stream>>>(xn1, DIMC, wv, kv_b + DIMC, nullptr,
                                                  nullptr, vbf, tiles, DIMC / 32);

  vsum_kernel<<<1, 256, 0, stream>>>(vbf, vs);
  packv_kernel<<<4096, 256, 0, stream>>>(vbf, vpk);

  // sparse attention
  attn_kernel<<<dim3(NTOK / 16, HEADS), 256, 0, stream>>>(qbf, kbf, vpk, vs, attnout);

  // output projection + residual(x) -> x1 (fp32)
  gemm16_kernel<0><<<tiles / 4, 128, 0, stream>>>(attnout, DIMC, wp, proj_b, x, x1,
                                                  nullptr, tiles, DIMC / 32);

  // LN2 -> channel-major fp32 for convs
  ln_kernel<<<NTOK, 256, 0, stream>>>(x1, ln2_g, ln2_b, nullptr, xn2t);

  // depthwise 3x3 + 5x5 -> xc (pair-interleaved, KS=4096)
  msfn_conv_kernel<<<(HID * NTOK) / 256, 256, 0, stream>>>(xn2t, c3_w, c3_b, c5_w,
                                                           c5_b, xc);

  // 1x1 conv as GEMM 4096x2048x256 + bias + residual(x1) -> out fp32
  gemm16_kernel<1><<<tiles / 4, 128, 0, stream>>>(xc, 0, c1wT, c1_b, x1, out, nullptr,
                                                  tiles, (2 * HID) / 32);
}
